// PointPillars_80556406604564
// MI455X (gfx1250) — matrix operand order repaired and checked by reading, not compiled, not run on hardware
//
#include <hip/hip_runtime.h>

typedef __attribute__((ext_vector_type(16))) _Float16 v16h;
typedef __attribute__((ext_vector_type(8)))  _Float16 h8;
typedef __attribute__((ext_vector_type(8)))  float    v8f;
typedef _Float16 half_t;

#define CAT16(a, b)                                                            \
  __builtin_shufflevector(a, b, 0, 1, 2, 3, 4, 5, 6, 7, 8, 9, 10, 11, 12, 13, \
                          14, 15)
#define WMMA_F16(A, Bv, C)                                                     \
  __builtin_amdgcn_wmma_f32_16x16x32_f16(false, A, false, Bv, (short)0, C,     \
                                         false, false)

// ----------------------------------------------------------------------------
// utility kernels
// ----------------------------------------------------------------------------
__global__ void fill_zero_u32(unsigned* __restrict__ p, long n) {
  long i = (long)blockIdx.x * blockDim.x + threadIdx.x;
  long stride = (long)gridDim.x * blockDim.x;
  for (; i < n; i += stride) p[i] = 0u;
}

// conv weight OIHW (Cout,Cin,3,3) f32 -> f16 [o][rs*Cin + ci]  (K = rs-major)
__global__ void conv_w_to_f16(const float* __restrict__ w, half_t* __restrict__ wq,
                              int Cout, int Cin) {
  int K = Cin * 9;
  long total = (long)Cout * K;
  long i = (long)blockIdx.x * blockDim.x + threadIdx.x;
  if (i >= total) return;
  int o = (int)(i / K);
  int t = (int)(i - (long)o * K);
  int rs = t / Cin, ci = t - rs * Cin;
  wq[i] = (half_t)w[((long)o * Cin + ci) * 9 + rs];
}

// deconv weight (Cin,Cout,k,k) f32 -> f16 [o][ij][Cin]
__global__ void deconv_w_to_f16(const float* __restrict__ w, half_t* __restrict__ wq,
                                int Cin, int Cout, int k) {
  long total = (long)Cout * k * k * Cin;
  long i = (long)blockIdx.x * blockDim.x + threadIdx.x;
  if (i >= total) return;
  int c = (int)(i % Cin);
  long rest = i / Cin;
  int ij = (int)(rest % (k * k));
  int o = (int)(rest / (k * k));
  int ii = ij / k, jj = ij - ii * k;
  wq[i] = (half_t)w[(((long)c * Cout + o) * k + ii) * k + jj];
}

// heads: cls(2,384), reg(14,384), dir(4,384) -> f16 [32][384] (rows 20..31 zero)
__global__ void head_w_to_f16(const float* __restrict__ cw, const float* __restrict__ rw,
                              const float* __restrict__ dw, const float* __restrict__ cb,
                              const float* __restrict__ rb, const float* __restrict__ db,
                              half_t* __restrict__ wq, float* __restrict__ hb) {
  int i = blockIdx.x * blockDim.x + threadIdx.x;
  if (i < 32) {
    float b = 0.f;
    if (i < 2) b = cb[i];
    else if (i < 16) b = rb[i - 2];
    else if (i < 20) b = db[i - 16];
    hb[i] = b;
  }
  if (i >= 32 * 384) return;
  int n = i / 384, k = i - n * 384;
  float v = 0.f;
  if (n < 2) v = cw[n * 384 + k];
  else if (n < 16) v = rw[(n - 2) * 384 + k];
  else if (n < 20) v = dw[(n - 16) * 384 + k];
  wq[i] = (half_t)v;
}

// ----------------------------------------------------------------------------
// pillar encoder -> NHWC canvas (B, Y=256, X=192, 64); 64 threads write the
// pillar's 64 channels contiguously (coalesced u16 stores)
// ----------------------------------------------------------------------------
__global__ __launch_bounds__(64)
void pillar_encode(const float* __restrict__ pts, const int* __restrict__ coors,
                   const int* __restrict__ npts, const float* __restrict__ pw,
                   const float* __restrict__ psc, const float* __restrict__ pbi,
                   half_t* __restrict__ canvas, int P) {
  int p = blockIdx.x;
  if (p >= P) return;
  int o = threadIdx.x;
  int np = npts[p];
  const float* q = pts + (long)p * 32 * 4;
  float sx = 0.f, sy = 0.f, sz = 0.f;  // padded points are zeroed in the input
  for (int i = 0; i < 32; ++i) {
    sx += q[i * 4 + 0]; sy += q[i * 4 + 1]; sz += q[i * 4 + 2];
  }
  float inv = 1.0f / (float)np;
  float mx = sx * inv, my = sy * inv, mz = sz * inv;
  float cx = (float)coors[p * 4 + 1] * 0.16f + 0.08f;   // VX/2 + 0.0
  float cy = (float)coors[p * 4 + 2] * 0.16f - 20.40f;  // VY/2 - 20.48
  const float* w = pw + o * 6;
  float w0 = w[0], w1 = w[1], w2 = w[2], w3 = w[3], w4 = w[4], w5 = w[5];
  float sc = psc[o], bi = pbi[o];
  float hmax = -3.0e38f;
  for (int i = 0; i < 32; ++i) {
    float x = q[i * 4 + 0], y = q[i * 4 + 1], z = q[i * 4 + 2];
    float h = 0.f;  // masked rows: features are zero
    if (i < np)
      h = z * w0 + (x - mx) * w1 + (y - my) * w2 + (z - mz) * w3 +
          (x - cx) * w4 + (y - cy) * w5;
    h = h * sc + bi;
    h = h > 0.f ? h : 0.f;
    hmax = hmax > h ? hmax : h;
  }
  int b = coors[p * 4 + 0], gx = coors[p * 4 + 1], gy = coors[p * 4 + 2];
  canvas[(((long)b * 256 + gy) * 192 + gx) * 64 + o] = (half_t)hmax;
}

// ----------------------------------------------------------------------------
// implicit-GEMM 3x3 conv (pad 1) + BN + ReLU, NHWC activations.
// One wave -> 32(M) x 32(N) tile: 4 accumulators share 2 A + 2 B fragments.
// K ordering: (r,s)-outer, channel-inner -> every fragment is 2 x b128 loads.
// Fast path per tap when all 32 pixels are interior (wave-uniform branch).
// ----------------------------------------------------------------------------
__global__ __launch_bounds__(256)
void conv3x3_bn_relu_wmma(const half_t* __restrict__ in, const half_t* __restrict__ wt,
                          const float* __restrict__ scale, const float* __restrict__ bias,
                          half_t* __restrict__ out, int B, int Cin, int Hin, int Win,
                          int Cout, int Hout, int Wout, int stride) {
  const int lane = threadIdx.x & 31;
  const int wid = blockIdx.x * (blockDim.x >> 5) + (threadIdx.x >> 5);
  const int numN = Cout >> 5;  // 32-wide N tiles
  const int HWo = Hout * Wout;
  const int M = B * HWo;
  const int numTiles = (M >> 5) * numN;
  if (wid >= numTiles) return;  // wave-uniform
  const int mTile = wid / numN, nTile = wid - mTile * numN;
  const int m0 = mTile << 5, n0 = nTile << 5;
  const int laneHi = lane >> 4, lm = lane & 15;
  // the two output pixels this lane contributes to the A fragments
  const int mA = m0 + lm, mB = m0 + 16 + lm;
  int bA = mA / HWo, pA = mA - bA * HWo;
  int oyA = pA / Wout, oxA = pA - oyA * Wout;
  int bB = mB / HWo, pB = mB - bB * HWo;
  int oyB = pB / Wout, oxB = pB - oyB * Wout;
  const int n = n0 + lm;  // low N column; high column is n+16
  const int K = Cin * 9;
  const half_t* wrow0 = wt + (long)n * K;         // [Cout][K]
  const half_t* wrow1 = wt + (long)(n + 16) * K;
  v8f acc00 = {}, acc01 = {}, acc10 = {}, acc11 = {};
  for (int r = 0; r < 3; ++r) {
    const int iyA = oyA * stride + r - 1;
    const int iyB = oyB * stride + r - 1;
    for (int s = 0; s < 3; ++s) {
      const int ixA = oxA * stride + s - 1;
      const int ixB = oxB * stride + s - 1;
      const bool okA = (unsigned)iyA < (unsigned)Hin && (unsigned)ixA < (unsigned)Win;
      const bool okB = (unsigned)iyB < (unsigned)Hin && (unsigned)ixB < (unsigned)Win;
      const half_t* pa =
          in + (((long)bA * Hin + iyA) * Win + ixA) * Cin + laneHi * 8;
      const half_t* pb =
          in + (((long)bB * Hin + iyB) * Win + ixB) * Cin + laneHi * 8;
      const int rsC = (r * 3 + s) * Cin + laneHi * 8;
      const half_t* pw0 = wrow0 + rsC;
      const half_t* pw1 = wrow1 + rsC;
      if (__all(okA && okB)) {
        // interior fast path: no masks, pure b128 streams
        for (int cb = 0; cb < Cin; cb += 32) {
          h8 a0 = *(const h8*)(pa + cb);
          h8 a1 = *(const h8*)(pa + cb + 16);
          h8 c0 = *(const h8*)(pb + cb);
          h8 c1 = *(const h8*)(pb + cb + 16);
          h8 w00 = *(const h8*)(pw0 + cb);
          h8 w01 = *(const h8*)(pw0 + cb + 16);
          h8 w10 = *(const h8*)(pw1 + cb);
          h8 w11 = *(const h8*)(pw1 + cb + 16);
          v16h av = CAT16(a0, a1), cv2 = CAT16(c0, c1);
          v16h bv0 = CAT16(w00, w01), bv1 = CAT16(w10, w11);
          acc00 = WMMA_F16(av, bv0, acc00);
          acc01 = WMMA_F16(av, bv1, acc01);
          acc10 = WMMA_F16(cv2, bv0, acc10);
          acc11 = WMMA_F16(cv2, bv1, acc11);
        }
      } else {
        for (int cb = 0; cb < Cin; cb += 32) {
          h8 a0 = {}, a1 = {}, c0 = {}, c1 = {};
          if (okA) {
            a0 = *(const h8*)(pa + cb);
            a1 = *(const h8*)(pa + cb + 16);
          }
          if (okB) {
            c0 = *(const h8*)(pb + cb);
            c1 = *(const h8*)(pb + cb + 16);
          }
          h8 w00 = *(const h8*)(pw0 + cb);
          h8 w01 = *(const h8*)(pw0 + cb + 16);
          h8 w10 = *(const h8*)(pw1 + cb);
          h8 w11 = *(const h8*)(pw1 + cb + 16);
          v16h av = CAT16(a0, a1), cv2 = CAT16(c0, c1);
          v16h bv0 = CAT16(w00, w01), bv1 = CAT16(w10, w11);
          acc00 = WMMA_F16(av, bv0, acc00);
          acc01 = WMMA_F16(av, bv1, acc01);
          acc10 = WMMA_F16(cv2, bv0, acc10);
          acc11 = WMMA_F16(cv2, bv1, acc11);
        }
      }
    }
  }
  const float sc0 = scale[n], bi0 = bias[n];
  const float sc1 = scale[n + 16], bi1 = bias[n + 16];
#pragma unroll
  for (int r = 0; r < 8; ++r) {  // D: M = r + laneHi*8, N = lane&15
    long mm0 = m0 + (laneHi << 3) + r;  // flat NHWC pixel == GEMM row
    long mm1 = mm0 + 16;
    float v00 = acc00[r] * sc0 + bi0, v01 = acc01[r] * sc1 + bi1;
    float v10 = acc10[r] * sc0 + bi0, v11 = acc11[r] * sc1 + bi1;
    v00 = v00 > 0.f ? v00 : 0.f;
    v01 = v01 > 0.f ? v01 : 0.f;
    v10 = v10 > 0.f ? v10 : 0.f;
    v11 = v11 > 0.f ? v11 : 0.f;
    out[mm0 * Cout + n] = (half_t)v00;
    out[mm0 * Cout + n + 16] = (half_t)v01;
    out[mm1 * Cout + n] = (half_t)v10;
    out[mm1 * Cout + n + 16] = (half_t)v11;
  }
}

// ----------------------------------------------------------------------------
// deconv (kernel==stride, incl. k=1 == 1x1 conv) + BN + ReLU, Cout=128,
// NHWC in, writes into NHWC feat (B,128,96,384) at channel offset chOff.
// blockIdx.y enumerates the (i,j) sub-position -> pure GEMM per sub-position.
// One wave -> 32(M) x 32(N).
// ----------------------------------------------------------------------------
__global__ __launch_bounds__(256)
void deconv_bn_relu_wmma(const half_t* __restrict__ in, const half_t* __restrict__ wq,
                         const float* __restrict__ scale, const float* __restrict__ bias,
                         half_t* __restrict__ feat, int B, int Cin, int Hin, int Win,
                         int k, int chOff) {
  const int Cout = 128;
  const int ij = blockIdx.y;
  const int ii = ij / k, jj = ij - ii * k;
  const int lane = threadIdx.x & 31;
  const int wid = blockIdx.x * (blockDim.x >> 5) + (threadIdx.x >> 5);
  const int numN = Cout >> 5;  // 4
  const int HWi = Hin * Win;
  const int M = B * HWi;
  const int numTiles = (M >> 5) * numN;
  if (wid >= numTiles) return;
  const int mTile = wid / numN, nTile = wid - mTile * numN;
  const int m0 = mTile << 5, n0 = nTile << 5;
  const int laneHi = lane >> 4, lm = lane & 15;
  const int n = n0 + lm;
  const half_t* pa = in + (long)(m0 + lm) * Cin + laneHi * 8;  // contiguous K
  const half_t* pb = in + (long)(m0 + 16 + lm) * Cin + laneHi * 8;
  const long wstep = (long)k * k * Cin;
  const half_t* pw0 = wq + (long)n * wstep + (long)ij * Cin + laneHi * 8;
  const half_t* pw1 = pw0 + 16 * wstep;
  v8f acc00 = {}, acc01 = {}, acc10 = {}, acc11 = {};
  for (int cb = 0; cb < Cin; cb += 32) {
    h8 a0 = *(const h8*)(pa + cb);
    h8 a1 = *(const h8*)(pa + cb + 16);
    h8 c0 = *(const h8*)(pb + cb);
    h8 c1 = *(const h8*)(pb + cb + 16);
    h8 w00 = *(const h8*)(pw0 + cb);
    h8 w01 = *(const h8*)(pw0 + cb + 16);
    h8 w10 = *(const h8*)(pw1 + cb);
    h8 w11 = *(const h8*)(pw1 + cb + 16);
    v16h av = CAT16(a0, a1), cv2 = CAT16(c0, c1);
    v16h bv0 = CAT16(w00, w01), bv1 = CAT16(w10, w11);
    acc00 = WMMA_F16(av, bv0, acc00);
    acc01 = WMMA_F16(av, bv1, acc01);
    acc10 = WMMA_F16(cv2, bv0, acc10);
    acc11 = WMMA_F16(cv2, bv1, acc11);
  }
  const float sc0 = scale[n], bi0 = bias[n];
  const float sc1 = scale[n + 16], bi1 = bias[n + 16];
#pragma unroll
  for (int r = 0; r < 8; ++r) {
#pragma unroll
    for (int halfM = 0; halfM < 2; ++halfM) {
      int mm = m0 + (laneHi << 3) + r + halfM * 16;
      int bb = mm / HWi;
      int pp = mm - bb * HWi;
      int hh = pp / Win, ww = pp - hh * Win;
      int oy = hh * k + ii, ox = ww * k + jj;
      long obase = (((long)bb * 128 + oy) * 96 + ox) * 384 + chOff;
      float v0 = (halfM ? acc10[r] : acc00[r]) * sc0 + bi0;
      float v1 = (halfM ? acc11[r] : acc01[r]) * sc1 + bi1;
      v0 = v0 > 0.f ? v0 : 0.f;
      v1 = v1 > 0.f ? v1 : 0.f;
      feat[obase + n] = (half_t)v0;
      feat[obase + n + 16] = (half_t)v1;
    }
  }
}

// ----------------------------------------------------------------------------
// heads: GEMM M = B*128*96, K = 384, N = 32 (cls 0..1, reg 2..15, dir 16..19)
// feat is NHWC so A fragments are contiguous. One wave -> 32(M) x 32(N).
// ----------------------------------------------------------------------------
__global__ __launch_bounds__(256)
void head_wmma(const half_t* __restrict__ feat, const half_t* __restrict__ wq,
               const float* __restrict__ hb, float* __restrict__ out, int B) {
  const int HW = 128 * 96;
  const int lane = threadIdx.x & 31;
  const int wid = blockIdx.x * (blockDim.x >> 5) + (threadIdx.x >> 5);
  const int M = B * HW;
  const int numTiles = M >> 5;  // single 32-wide N tile
  if (wid >= numTiles) return;
  const int m0 = wid << 5;
  const int laneHi = lane >> 4, lm = lane & 15;
  const int n = lm;
  const half_t* pa = feat + (long)(m0 + lm) * 384 + laneHi * 8;
  const half_t* pb = feat + (long)(m0 + 16 + lm) * 384 + laneHi * 8;
  const half_t* pw0 = wq + (long)n * 384 + laneHi * 8;  // [32][384]
  const half_t* pw1 = pw0 + 16 * 384;
  v8f acc00 = {}, acc01 = {}, acc10 = {}, acc11 = {};
  for (int cb = 0; cb < 384; cb += 32) {
    h8 a0 = *(const h8*)(pa + cb);
    h8 a1 = *(const h8*)(pa + cb + 16);
    h8 c0 = *(const h8*)(pb + cb);
    h8 c1 = *(const h8*)(pb + cb + 16);
    h8 w00 = *(const h8*)(pw0 + cb);
    h8 w01 = *(const h8*)(pw0 + cb + 16);
    h8 w10 = *(const h8*)(pw1 + cb);
    h8 w11 = *(const h8*)(pw1 + cb + 16);
    v16h av = CAT16(a0, a1), cv2 = CAT16(c0, c1);
    v16h bv0 = CAT16(w00, w01), bv1 = CAT16(w10, w11);
    acc00 = WMMA_F16(av, bv0, acc00);
    acc01 = WMMA_F16(av, bv1, acc01);
    acc10 = WMMA_F16(cv2, bv0, acc10);
    acc11 = WMMA_F16(cv2, bv1, acc11);
  }
  const float bi0 = hb[n], bi1 = hb[n + 16];
  const int regOff = B * 2 * HW;
  const int dirOff = regOff + B * 14 * HW;
#pragma unroll
  for (int r = 0; r < 8; ++r) {
#pragma unroll
    for (int halfM = 0; halfM < 2; ++halfM) {
      int mm = m0 + (laneHi << 3) + r + halfM * 16;
      int bb = mm / HW;
      int pp = mm - bb * HW;
      // low N column (n in 0..15): cls or reg
      {
        float v = (halfM ? acc10[r] : acc00[r]) + bi0;
        int idx = (n < 2) ? (bb * 2 + n) * HW + pp
                          : regOff + (bb * 14 + (n - 2)) * HW + pp;
        out[idx] = v;
      }
      // high N column (n+16 in 16..31): dir (16..19) or padding (skip)
      if (n < 4) {
        float v = (halfM ? acc11[r] : acc01[r]) + bi1;
        out[dirOff + (bb * 4 + n) * HW + pp] = v;
      }
    }
  }
}

// ----------------------------------------------------------------------------
// host orchestration
// ----------------------------------------------------------------------------
extern "C" void kernel_launch(void* const* d_in, const int* in_sizes, int n_in,
                              void* d_out, int out_size, void* d_ws, size_t ws_size,
                              hipStream_t stream) {
  (void)n_in; (void)ws_size;
  // ---- input pointers (setup_inputs() dict order, pytree-leaf flattened) ----
  const float* pillars = (const float*)d_in[0];
  const int*   coors   = (const int*)d_in[1];
  const int*   npts    = (const int*)d_in[2];
  // d_in[3] = batch_size scalar (derived from out_size instead)
  const float* pe_w = (const float*)d_in[4];
  const float* pe_s = (const float*)d_in[5];
  const float* pe_b = (const float*)d_in[6];
  const float *bbw[16], *bbs[16], *bbb[16];
  for (int i = 0; i < 16; ++i) {
    bbw[i] = (const float*)d_in[7 + i];
    bbs[i] = (const float*)d_in[23 + i];
    bbb[i] = (const float*)d_in[39 + i];
  }
  const float *nw[3], *nsc[3], *nbi[3];
  for (int i = 0; i < 3; ++i) {
    nw[i]  = (const float*)d_in[55 + i];
    nsc[i] = (const float*)d_in[58 + i];
    nbi[i] = (const float*)d_in[61 + i];
  }
  const float* cw  = (const float*)d_in[64];
  const float* cb  = (const float*)d_in[65];
  const float* rw  = (const float*)d_in[66];
  const float* rb  = (const float*)d_in[67];
  const float* dwt = (const float*)d_in[68];
  const float* db  = (const float*)d_in[69];

  const int P = in_sizes[0] / (32 * 4);
  const int B = out_size / (20 * 128 * 96);  // 4

  // ---- workspace layout ----
  char* ws = (char*)d_ws;
  size_t cur = 0;
  auto alloc = [&](size_t bytes) -> void* {
    void* p = ws + cur;
    cur = (cur + bytes + 255) & ~(size_t)255;
    return p;
  };
  half_t* cv  = (half_t*)alloc((size_t)B * 256 * 192 * 64 * sizeof(half_t));  // NHWC canvas
  half_t* pp0 = (half_t*)alloc((size_t)B * 128 * 96 * 64 * sizeof(half_t));
  half_t* pp1 = (half_t*)alloc((size_t)B * 128 * 96 * 64 * sizeof(half_t));
  half_t* o1  = (half_t*)alloc((size_t)B * 128 * 96 * 64 * sizeof(half_t));
  half_t* o2  = (half_t*)alloc((size_t)B * 64 * 48 * 128 * sizeof(half_t));
  half_t* o3  = (half_t*)alloc((size_t)B * 32 * 24 * 256 * sizeof(half_t));
  half_t* feat = (half_t*)alloc((size_t)B * 128 * 96 * 384 * sizeof(half_t));

  struct Cfg { int Cin, Cout, Hin, Win, stride; };
  const Cfg cfg[16] = {
    {64, 64, 256, 192, 2}, {64, 64, 128, 96, 1}, {64, 64, 128, 96, 1}, {64, 64, 128, 96, 1},
    {64, 128, 128, 96, 2}, {128, 128, 64, 48, 1}, {128, 128, 64, 48, 1},
    {128, 128, 64, 48, 1}, {128, 128, 64, 48, 1}, {128, 128, 64, 48, 1},
    {128, 256, 64, 48, 2}, {256, 256, 32, 24, 1}, {256, 256, 32, 24, 1},
    {256, 256, 32, 24, 1}, {256, 256, 32, 24, 1}, {256, 256, 32, 24, 1}};
  half_t* wq[16];
  for (int i = 0; i < 16; ++i)
    wq[i] = (half_t*)alloc((size_t)cfg[i].Cin * 9 * cfg[i].Cout * sizeof(half_t));
  const int nk[3] = {1, 2, 4};
  const int nCin[3] = {64, 128, 256};
  half_t* nq[3];
  for (int i = 0; i < 3; ++i)
    nq[i] = (half_t*)alloc((size_t)nk[i] * nk[i] * nCin[i] * 128 * sizeof(half_t));
  half_t* hq = (half_t*)alloc(32 * 384 * sizeof(half_t));
  float*  hb = (float*)alloc(32 * sizeof(float));

  // ---- weight packing (tiny, every call: deterministic & graph-safe) ----
  for (int i = 0; i < 16; ++i) {
    long total = (long)cfg[i].Cin * 9 * cfg[i].Cout;
    conv_w_to_f16<<<(int)((total + 255) / 256), 256, 0, stream>>>(
        bbw[i], wq[i], cfg[i].Cout, cfg[i].Cin);
  }
  for (int i = 0; i < 3; ++i) {
    long total = (long)nk[i] * nk[i] * nCin[i] * 128;
    deconv_w_to_f16<<<(int)((total + 255) / 256), 256, 0, stream>>>(
        nw[i], nq[i], nCin[i], 128, nk[i]);
  }
  head_w_to_f16<<<(32 * 384 + 255) / 256, 256, 0, stream>>>(cw, rw, dwt, cb, rb, db,
                                                            hq, hb);

  // ---- pillar encoder + scatter ----
  long cvU32 = (long)B * 256 * 192 * 64 / 2;
  fill_zero_u32<<<4096, 256, 0, stream>>>((unsigned*)cv, cvU32);
  pillar_encode<<<P, 64, 0, stream>>>(pillars, coors, npts, pe_w, pe_s, pe_b, cv, P);

  // ---- backbone: 16 convs (ping-pong; last of each block -> o1/o2/o3) ----
  const half_t* inBuf[16]  = {cv,  pp0, pp1, pp0, o1,  pp0, pp1, pp0,
                              pp1, pp0, o2,  pp0, pp1, pp0, pp1, pp0};
  half_t*       outBuf[16] = {pp0, pp1, pp0, o1,  pp0, pp1, pp0, pp1,
                              pp0, o2,  pp0, pp1, pp0, pp1, pp0, o3};
  for (int i = 0; i < 16; ++i) {
    const Cfg& c = cfg[i];
    int Hout = (c.Hin - 1) / c.stride + 1;
    int Wout = (c.Win - 1) / c.stride + 1;
    long waves = ((long)B * Hout * Wout / 32) * (c.Cout / 32);
    int blocks = (int)((waves + 7) / 8);
    conv3x3_bn_relu_wmma<<<blocks, 256, 0, stream>>>(
        inBuf[i], wq[i], bbs[i], bbb[i], outBuf[i], B, c.Cin, c.Hin, c.Win, c.Cout,
        Hout, Wout, c.stride);
  }

  // ---- neck: writes all 384 channels of feat (full coverage, no pre-zero) ----
  const half_t* nin[3] = {o1, o2, o3};
  const int nH[3] = {128, 64, 32}, nW[3] = {96, 48, 24};
  for (int i = 0; i < 3; ++i) {
    long waves = ((long)B * nH[i] * nW[i] / 32) * (128 / 32);
    dim3 grid((unsigned)((waves + 7) / 8), (unsigned)(nk[i] * nk[i]));
    deconv_bn_relu_wmma<<<grid, 256, 0, stream>>>(nin[i], nq[i], nsc[i], nbi[i], feat,
                                                  B, nCin[i], nH[i], nW[i], nk[i],
                                                  128 * i);
  }

  // ---- heads -> d_out (cls | reg | dir), f32 ----
  {
    long waves = (long)B * 128 * 96 / 32;
    int blocks = (int)((waves + 7) / 8);
    head_wmma<<<blocks, 256, 0, stream>>>(feat, hq, hb, (float*)d_out, B);
  }
}
